// GraphDecoderHomo_86122684219720
// MI455X (gfx1250) — matrix operand order repaired and checked by reading, not compiled, not run on hardware
//
#include <hip/hip_runtime.h>
#include <cstdint>
#include <cstddef>

// ---------------------------------------------------------------------------
// GCN decoder: 3 x (src-norm -> GEMM -> edge gather*w -> scatter-sum -> dst-norm
// + bias [+ReLU]).  GEMMs for 256->128 and 128->64 use v_wmma_f32_16x16x32_bf16
// (bf16 A/B, f32 accumulate).  Edge phase uses coalesced float4 gathers and
// native f32 global atomics.  Layer 3 (64->1) is a scalar dot.
// ---------------------------------------------------------------------------

typedef __attribute__((ext_vector_type(16))) __bf16 v16bf;
typedef __attribute__((ext_vector_type(8)))  float  v8f;

#define DIM0 256
#define DIM1 128
#define DIM2 64

__device__ __forceinline__ void atomic_add_f32(float* p, float v) {
  unsafeAtomicAdd(p, v);   // -> global_atomic_add_f32
}

__device__ __forceinline__ unsigned short bf16_bits(float x) {
  __bf16 h = (__bf16)x;
  return __builtin_bit_cast(unsigned short, h);
}

// -------------------------------- utility ----------------------------------
__global__ void k_zero_f32(float* __restrict__ p, int n) {
  int i = blockIdx.x * blockDim.x + threadIdx.x;
  if (i < n) p[i] = 0.0f;
}

__global__ void k_degrees(const int* __restrict__ src, const int* __restrict__ dst,
                          float* __restrict__ degO, float* __restrict__ degI, int E) {
  int e = blockIdx.x * blockDim.x + threadIdx.x;
  if (e < E) {
    atomic_add_f32(degO + src[e], 1.0f);
    atomic_add_f32(degI + dst[e], 1.0f);
  }
}

__global__ void k_rsqrt_clamp(float* __restrict__ a, int n) {
  int i = blockIdx.x * blockDim.x + threadIdx.x;
  if (i < n) a[i] = rsqrtf(fmaxf(a[i], 1.0f));
}

// Pre-pack W [K,OUT] f32 row-major into the per-lane bf16 B-matrix layout for
// v_wmma_f32_16x16x32_bf16:
//   dword index = ((kc*NT + t)*32 + lane)*8 + j
//   lane = (lo,hi): column n = t*16+lo, K-half = hi*16, dword j = K {2j, 2j+1}
__global__ void k_pack_w(const float* __restrict__ W, uint32_t* __restrict__ wp,
                         int OUT, int total) {
  int idx = blockIdx.x * blockDim.x + threadIdx.x;
  if (idx >= total) return;
  int j    = idx & 7;
  int lane = (idx >> 3) & 31;
  int tile = idx >> 8;              // kc*NT + t
  int NT   = OUT >> 4;
  int kc   = tile / NT;
  int t    = tile - kc * NT;
  int lo = lane & 15, hi = lane >> 4;
  int n  = t * 16 + lo;
  int k0 = kc * 32 + hi * 16 + 2 * j;
  unsigned short u0 = bf16_bits(W[(size_t)k0 * OUT + n]);
  unsigned short u1 = bf16_bits(W[(size_t)(k0 + 1) * OUT + n]);
  wp[idx] = (uint32_t)u0 | ((uint32_t)u1 << 16);
}

// ------------------------------ WMMA GEMM ----------------------------------
// H[N,OUT] = (Z[N,K] * norm_src[:,None]) @ Wpack   (bf16 x bf16 -> f32)
// One wave per 16-row M-tile; 4 waves per block.
template <int K, int OUT>
__global__ __launch_bounds__(128)
void k_gemm_wmma(const float* __restrict__ Z, const float* __restrict__ norm_src,
                 const uint32_t* __restrict__ Wpack, float* __restrict__ H, int nrows) {
  constexpr int NT = OUT / 16;   // N tiles
  constexpr int KC = K / 32;     // K chunks
  const int wave = threadIdx.x >> 5;
  const int lane = threadIdx.x & 31;
  const int m_tile = blockIdx.x * 4 + wave;
  const int m_base = m_tile * 16;
  if (m_base >= nrows) return;            // wave-uniform: EXEC stays all-ones
  const int lo = lane & 15, hi = lane >> 4;
  const int row = m_base + lo;            // both K-halves of a row share `row`
  const float ns = norm_src[row];
  const float* zrow = Z + (size_t)row * K;

  v8f acc[NT];
#pragma unroll
  for (int t = 0; t < NT; t++)
#pragma unroll
    for (int r = 0; r < 8; r++) acc[t][r] = 0.0f;

#pragma unroll
  for (int kc = 0; kc < KC; kc++) {
    // A (16x32 bf16): lane(lo,hi) -> M=lo, K = kc*32 + hi*16 + {0..15}
    v16bf a;
    const float* p0 = zrow + kc * 32 + hi * 16;
    const float* p1 = p0 + 8;
#pragma unroll
    for (int j = 0; j < 8; j++) a[j]     = (__bf16)(p0[j] * ns);
#pragma unroll
    for (int j = 0; j < 8; j++) a[8 + j] = (__bf16)(p1[j] * ns);

#pragma unroll
    for (int t = 0; t < NT; t++) {
      const uint32_t* bp = Wpack + ((size_t)(kc * NT + t) * 32 + lane) * 8;
      v16bf b = *(const v16bf*)bp;        // 32B contiguous per lane, coalesced
      acc[t] = __builtin_amdgcn_wmma_f32_16x16x32_bf16(
          /*neg_a=*/false, a, /*neg_b=*/false, b,
          /*c_mod=*/(short)0, acc[t], /*reuse_a=*/false, /*reuse_b=*/false);
    }
  }

  // C/D layout: lane(lo,hi), VGPR r -> M = hi*8 + r, N = lo
#pragma unroll
  for (int t = 0; t < NT; t++)
#pragma unroll
    for (int r = 0; r < 8; r++) {
      int m = m_base + hi * 8 + r;
      if (m < nrows) H[(size_t)m * OUT + t * 16 + lo] = acc[t][r];
    }
}

// ------------------------------ edge phase ---------------------------------
template <int OUT>
__global__ void k_scatter(const float* __restrict__ H, const int* __restrict__ src,
                          const int* __restrict__ dst, const float* __restrict__ ew,
                          float* __restrict__ R, int E) {
  constexpr int C = OUT / 4;              // float4 chunks per edge
  int idx = blockIdx.x * blockDim.x + threadIdx.x;
  if (idx >= E * C) return;
  int e = idx / C;
  int c = idx - e * C;
  int s = src[e], d = dst[e];
  float w = ew[e];
  const float4 hv = *(const float4*)(H + (size_t)s * OUT + c * 4);
  float* rp = R + (size_t)d * OUT + c * 4;
  atomic_add_f32(rp + 0, hv.x * w);
  atomic_add_f32(rp + 1, hv.y * w);
  atomic_add_f32(rp + 2, hv.z * w);
  atomic_add_f32(rp + 3, hv.w * w);
}

__global__ void k_scatter1(const float* __restrict__ H, const int* __restrict__ src,
                           const int* __restrict__ dst, const float* __restrict__ ew,
                           float* __restrict__ R, int E) {
  int e = blockIdx.x * blockDim.x + threadIdx.x;
  if (e < E) atomic_add_f32(R + dst[e], H[src[e]] * ew[e]);
}

__global__ void k_post(float* __restrict__ R, const float* __restrict__ normD,
                       const float* __restrict__ bias, int n, int OUT, int relu) {
  int i = blockIdx.x * blockDim.x + threadIdx.x;
  if (i >= n) return;
  int row = i / OUT;
  int col = i - row * OUT;
  float v = R[i] * normD[row] + bias[col];
  if (relu) v = fmaxf(v, 0.0f);
  R[i] = v;
}

// layer 3: h3[r] = norm_src[r] * dot(z3[r,:64], W3[:,0])
__global__ void k_dot64(const float* __restrict__ Z, const float* __restrict__ normS,
                        const float* __restrict__ W3, float* __restrict__ H, int N) {
  int r = blockIdx.x * blockDim.x + threadIdx.x;
  if (r >= N) return;
  const float* z = Z + (size_t)r * DIM2;
  float acc = 0.0f;
#pragma unroll
  for (int k = 0; k < DIM2; k++) acc += z[k] * W3[k];
  H[r] = acc * normS[r];
}

// ------------------------------- launcher ----------------------------------
extern "C" void kernel_launch(void* const* d_in, const int* in_sizes, int n_in,
                              void* d_out, int out_size, void* d_ws, size_t ws_size,
                              hipStream_t stream) {
  const float* b_z = (const float*)d_in[0];
  const int*   src = (const int*)d_in[1];
  const int*   dst = (const int*)d_in[2];
  const float* ew  = (const float*)d_in[3];
  const float* W1  = (const float*)d_in[5];
  const float* b1  = (const float*)d_in[6];
  const float* W2  = (const float*)d_in[7];
  const float* b2  = (const float*)d_in[8];
  const float* W3  = (const float*)d_in[9];
  const float* b3  = (const float*)d_in[10];

  const int N = in_sizes[0] / DIM0;       // 100000
  const int E = in_sizes[1];              // 1600000

  // workspace carve-out (256B aligned chunks)
  char* ws = (char*)d_ws;
  size_t off = 0;
  auto carve = [&](size_t bytes) -> void* {
    void* p = ws + off;
    off = (off + bytes + 255) & ~(size_t)255;
    return p;
  };
  float*    normS = (float*)carve((size_t)N * 4);
  float*    normD = (float*)carve((size_t)N * 4);
  float*    bufA  = (float*)carve((size_t)N * DIM1 * 4);   // h buffer
  float*    bufB  = (float*)carve((size_t)N * DIM1 * 4);   // rst / activations
  uint32_t* wp1   = (uint32_t*)carve((size_t)(DIM0 / 2) * DIM1 * 4);
  uint32_t* wp2   = (uint32_t*)carve((size_t)(DIM1 / 2) * DIM2 * 4);
  (void)ws_size; (void)n_in; (void)out_size;

  const int T = 256;
  auto nb = [&](long long n) { return (unsigned)((n + T - 1) / T); };
  const int m_tiles  = (N + 15) / 16;
  const unsigned gblk = (unsigned)((m_tiles + 3) / 4);

  // degrees -> norms (shared by all layers)
  k_zero_f32<<<nb(N), T, 0, stream>>>(normS, N);
  k_zero_f32<<<nb(N), T, 0, stream>>>(normD, N);
  k_degrees<<<nb(E), T, 0, stream>>>(src, dst, normS, normD, E);
  k_rsqrt_clamp<<<nb(N), T, 0, stream>>>(normS, N);
  k_rsqrt_clamp<<<nb(N), T, 0, stream>>>(normD, N);

  // pre-pack weights into WMMA B layout (bf16)
  k_pack_w<<<nb((DIM0 / 2) * DIM1), T, 0, stream>>>(W1, wp1, DIM1, (DIM0 / 2) * DIM1);
  k_pack_w<<<nb((DIM1 / 2) * DIM2), T, 0, stream>>>(W2, wp2, DIM2, (DIM1 / 2) * DIM2);

  // ---- layer 1: 256 -> 128 ----
  k_gemm_wmma<DIM0, DIM1><<<gblk, 128, 0, stream>>>(b_z, normS, wp1, bufA, N);
  k_zero_f32<<<nb((long long)N * DIM1), T, 0, stream>>>(bufB, N * DIM1);
  k_scatter<DIM1><<<nb((long long)E * (DIM1 / 4)), T, 0, stream>>>(bufA, src, dst, ew, bufB, E);
  k_post<<<nb((long long)N * DIM1), T, 0, stream>>>(bufB, normD, b1, N * DIM1, DIM1, 1);

  // ---- layer 2: 128 -> 64 ----
  k_gemm_wmma<DIM1, DIM2><<<gblk, 128, 0, stream>>>(bufB, normS, wp2, bufA, N);
  k_zero_f32<<<nb((long long)N * DIM2), T, 0, stream>>>(bufB, N * DIM2);
  k_scatter<DIM2><<<nb((long long)E * (DIM2 / 4)), T, 0, stream>>>(bufA, src, dst, ew, bufB, E);
  k_post<<<nb((long long)N * DIM2), T, 0, stream>>>(bufB, normD, b2, N * DIM2, DIM2, 1);

  // ---- layer 3: 64 -> 1 (scalar dot; WMMA with N=1 wastes the tile) ----
  k_dot64<<<nb(N), T, 0, stream>>>(bufB, normS, W3, bufA, N);
  float* outF = (float*)d_out;
  k_zero_f32<<<nb(N), T, 0, stream>>>(outF, N);
  k_scatter1<<<nb(E), T, 0, stream>>>(bufA, src, dst, ew, outF, E);
  k_post<<<nb(N), T, 0, stream>>>(outF, normD, b3, N, 1, 0);
}
